// ProtoModule_17678085390628
// MI455X (gfx1250) — matrix operand order repaired and checked by reading, not compiled, not run on hardware
//
#include <hip/hip_runtime.h>
#include <hip/hip_bf16.h>
#include <math.h>

// MI455X / gfx1250, wave32. GEMM core: v_wmma_f32_16x16x32_bf16.
typedef __attribute__((ext_vector_type(16))) __bf16 v16bf;
typedef __attribute__((ext_vector_type(8)))  float  v8f;

#define THRESH 0.95f

// ---------------- P0: binarize mask -> proto_grid output, valid flags ----------------
// 1 block, 256 threads; thread p handles one prototype's 8x8 mask block.
__global__ void k_mask(const float* __restrict__ sup_y,
                       float* __restrict__ out_grid,
                       unsigned* __restrict__ valid) {
  int p = threadIdx.x;                 // 0..255
  int b = p >> 6, y = (p >> 3) & 7, x = p & 7;
  int cnt = 0;
  for (int i = 0; i < 8; ++i)
    for (int j = 0; j < 8; ++j) {
      int idx = b * 4096 + (y * 8 + i) * 64 + (x * 8 + j);
      float bin = (sup_y[idx] > THRESH) ? 1.0f : 0.0f;
      cnt += (int)bin;
      out_grid[idx] = bin;
    }
  // mask_pooled > 0.5  <=>  sum of 64 binary values > 32
  valid[p] = (cnt > 32) ? 1u : 0u;
}

// ---------------- P1: 8x8 avg-pool of sup_x -> raw protos (f32, [p][c]) ----------------
// 1024 blocks (b*256+c), 64 threads. Coalesced row loads, shuffle reduce across w.
__global__ void k_pool(const float* __restrict__ sup_x,
                       float* __restrict__ protos_raw) {
  int b = blockIdx.x >> 8, c = blockIdx.x & 255;
  const float* plane = sup_x + (size_t)(b * 256 + c) * 4096;
  int t = threadIdx.x;                 // column w = t
  float cs[8] = {0, 0, 0, 0, 0, 0, 0, 0};
  for (int h = 0; h < 64; ++h)
    cs[h >> 3] += plane[h * 64 + t];   // 64 coalesced 256B row loads
  // reduce each h-block sum across the 8 lanes of each w-group
#pragma unroll
  for (int hp = 0; hp < 8; ++hp) {
    cs[hp] += __shfl_xor(cs[hp], 1);
    cs[hp] += __shfl_xor(cs[hp], 2);
    cs[hp] += __shfl_xor(cs[hp], 4);
  }
  float s = cs[0];
#pragma unroll
  for (int hp = 1; hp < 8; ++hp)
    if ((t & 7) == hp) s = cs[hp];
  int p = b * 64 + (t & 7) * 8 + (t >> 3);     // (b, h', w') -> flat proto id
  protos_raw[p * 256 + c] = s * 0.015625f;     // /64
}

// ---------------- P2: normalize protos, convert bf16, pre-swizzle to B-fragment order --
// 256 blocks (p), 256 threads (c). B-frag (t=N-tile, s=K-step): lane L holds
// proto[t*16 + (L&15)], channels s*32 + (L>>4)*8 + {0..7, 16..23}  (ISA 16-bit B 32x16 layout).
__global__ void k_norm(const float* __restrict__ protos_raw,
                       __bf16* __restrict__ protos_swz) {
  int p = blockIdx.x, c = threadIdx.x;
  float v = protos_raw[p * 256 + c];
  float ss = v * v;
  ss += __shfl_xor(ss, 1);
  ss += __shfl_xor(ss, 2);
  ss += __shfl_xor(ss, 4);
  ss += __shfl_xor(ss, 8);
  ss += __shfl_xor(ss, 16);
  __shared__ float red[8];
  if ((c & 31) == 0) red[c >> 5] = ss;
  __syncthreads();
  float tot = 0.f;
#pragma unroll
  for (int i = 0; i < 8; ++i) tot += red[i];
  float invn = 1.0f / fmaxf(sqrtf(tot), 1e-4f);
  __bf16 bv = (__bf16)(v * invn);
  int tt = p >> 4, nl = p & 15, s = c >> 5, half = (c >> 3) & 1, e = (c & 7) | ((c >> 1) & 8);
  // frag block (t,s) = 512 bf16; within it: lane*16 + element
  protos_swz[(size_t)((tt * 8 + s) * 512 + half * 256 + nl * 16 + e)] = bv;
}

// ---------------- Main: normalize-fold + WMMA GEMM + fused masked softmax --------------
// 1024 blocks x 256 threads (8 waves). Each wave owns 16 pixels x 256 protos.
__global__ __launch_bounds__(256) void k_main(const float* __restrict__ qry,
                                              const __bf16* __restrict__ protos_swz,
                                              const unsigned* __restrict__ valid,
                                              float* __restrict__ pred,
                                              float* __restrict__ assign_out) {
  const int lane = threadIdx.x & 31;
  const int wave = threadIdx.x >> 5;
  const int tile = blockIdx.x * 8 + wave;      // 8192 tiles of 16 pixels
  const int nl = lane & 15;
  const int half = lane >> 4;

  const int posb = tile * 16;                  // flat pixel index base (b*4096 + h*64 + w0)
  const int b = posb >> 12;
  const int hw = posb & 4095;
  const float* qbase = qry + (size_t)b * 1048576 + hw + nl;  // pixel row m = nl

  // A fragments (16-bit A 16x32 layout): lane row m=nl; element i -> K = i + (i>=8?8:0) + 8*half
  v16bf a[8];
  float sumsq = 0.f;
#pragma unroll
  for (int s = 0; s < 8; ++s) {
    const int c0 = s * 32 + half * 8;
#pragma unroll
    for (int j = 0; j < 8; ++j) {
      float v0 = __builtin_nontemporal_load(qbase + (size_t)(c0 + j) * 4096);
      float v1 = __builtin_nontemporal_load(qbase + (size_t)(c0 + 16 + j) * 4096);
      sumsq = fmaf(v0, v0, sumsq);
      sumsq = fmaf(v1, v1, sumsq);
      a[s][j]     = (__bf16)v0;
      a[s][j + 8] = (__bf16)v1;
    }
  }
  sumsq += __shfl_xor(sumsq, 16);              // combine the two K-halves of row nl
  // dist = (x . proto_n) * 20 / max(||x||, eps)  -- norm folded in post-GEMM (exact f32)
  float scale_m = 20.0f / fmaxf(sqrtf(sumsq), 1e-4f);

  // GEMM: 16 N-tiles x 8 K-steps = 128 WMMAs, A in regs, B streamed (swizzled, coalesced)
  v8f acc[16];
  v8f zero = {0.f, 0.f, 0.f, 0.f, 0.f, 0.f, 0.f, 0.f};
#pragma unroll
  for (int t = 0; t < 16; ++t) acc[t] = zero;
  const v16bf* bp = (const v16bf*)protos_swz;
#pragma unroll
  for (int t = 0; t < 16; ++t) {
#pragma unroll
    for (int s = 0; s < 8; ++s) {
      v16bf bf = bp[(t * 8 + s) * 32 + lane];  // 32B/lane, 1KB contiguous per frag
      acc[t] = __builtin_amdgcn_wmma_f32_16x16x32_bf16(
          false, a[s], false, bf, (short)0, acc[t], false, false);
    }
  }

  // valid bits for this lane's 16 columns (n = t*16 + nl)
  unsigned vm = 0;
#pragma unroll
  for (int t = 0; t < 16; ++t) vm |= (valid[t * 16 + nl] & 1u) << t;

  // D layout: lane holds n=nl, rows m = r + 8*half in acc[t][r]. Softmax over n per row.
#pragma unroll
  for (int r = 0; r < 8; ++r) {
    float sc = __shfl(scale_m, r + 8 * half);  // invnorm*20 for row m = r + 8*half
    float dv[16];
#pragma unroll
    for (int t = 0; t < 16; ++t) dv[t] = acc[t][r] * sc;

    // masked max + first-index argmax
    float vmax = -__builtin_inff();
    int vidx = nl;
#pragma unroll
    for (int t = 0; t < 16; ++t) {
      bool ok = (vm >> t) & 1u;
      if (ok && dv[t] > vmax) { vmax = dv[t]; vidx = t * 16 + nl; }
    }
#pragma unroll
    for (int msk = 1; msk <= 8; msk <<= 1) {
      float om = __shfl_xor(vmax, msk);
      int   oi = __shfl_xor(vidx, msk);
      if (om > vmax || (om == vmax && oi < vidx)) { vmax = om; vidx = oi; }
    }
    // exp-sum and weighted sum (invalid -> contributes 0, matches where(vmask,...,0))
    float se = 0.f, sed = 0.f;
#pragma unroll
    for (int t = 0; t < 16; ++t) {
      bool ok = (vm >> t) & 1u;
      float e = ok ? __expf(dv[t] - vmax) : 0.0f;
      se += e;
      sed = fmaf(e, dv[t], sed);
    }
#pragma unroll
    for (int msk = 1; msk <= 8; msk <<= 1) {
      se  += __shfl_xor(se, msk);
      sed += __shfl_xor(sed, msk);
    }
    if (nl == r) {                       // lanes r and r+16 write rows r and r+8
      int pos = posb + r + 8 * half;
      pred[pos] = sed / se;
      assign_out[pos] = (float)vidx;
    }
  }
}

extern "C" void kernel_launch(void* const* d_in, const int* in_sizes, int n_in,
                              void* d_out, int out_size, void* d_ws, size_t ws_size,
                              hipStream_t stream) {
  const float* qry   = (const float*)d_in[0];  // (32,256,64,64)
  const float* sup_x = (const float*)d_in[1];  // (4,256,64,64)
  const float* sup_y = (const float*)d_in[2];  // (4,1,64,64)

  float* out = (float*)d_out;
  float* out_pred   = out;            // 131072
  float* out_assign = out + 131072;   // 131072
  float* out_grid   = out + 262144;   // 16384

  char* ws = (char*)d_ws;
  float*    protos_raw = (float*)(ws);                 // 256*256*4 = 256 KB
  __bf16*   protos_swz = (__bf16*)(ws + 262144);       // 256*256*2 = 128 KB
  unsigned* valid      = (unsigned*)(ws + 262144 + 131072);  // 1 KB

  k_mask<<<1, 256, 0, stream>>>(sup_y, out_grid, valid);
  k_pool<<<1024, 64, 0, stream>>>(sup_x, protos_raw);
  k_norm<<<256, 256, 0, stream>>>(protos_raw, protos_swz);
  k_main<<<1024, 256, 0, stream>>>(qry, protos_swz, valid, out_pred, out_assign);
}
